// GCNDecoder_21887153340949
// MI455X (gfx1250) — matrix operand order repaired and checked by reading, not compiled, not run on hardware
//
#include <hip/hip_runtime.h>
#include <hip/hip_bf16.h>

#define NN   8192
#define HD   512
#define NLAY 3
#define LN_EPS 1e-5f

typedef __attribute__((ext_vector_type(16))) __bf16 v16bf;
typedef __attribute__((ext_vector_type(8)))  __bf16 v8bf;
typedef __attribute__((ext_vector_type(8)))  float  v8f;

// ---------------------------------------------------------------------------
// helpers
// ---------------------------------------------------------------------------
__device__ __forceinline__ void split_bf16(float x, __bf16& hi, __bf16& lo) {
    hi = (__bf16)x;
    lo = (__bf16)(x - (float)hi);
}

// A-operand fragment: 16x32 bf16, lane<16 holds K 0..15,
// lane>=16 holds K 16..31 of row (lane&15). Loaded from f32 row-major,
// split to hi/lo bf16 on the fly. RELU applied for the adjacency operand.
template <bool RELU>
__device__ __forceinline__ void load_a_frag(const float* __restrict__ rowptr,
                                            int k0, v16bf& ah, v16bf& al) {
    const int k1 = k0 + 8;
    float4 r0a = *(const float4*)(rowptr + k0);
    float4 r0b = *(const float4*)(rowptr + k0 + 4);
    float4 r1a = *(const float4*)(rowptr + k1);
    float4 r1b = *(const float4*)(rowptr + k1 + 4);
    float v[16] = {r0a.x, r0a.y, r0a.z, r0a.w, r0b.x, r0b.y, r0b.z, r0b.w,
                   r1a.x, r1a.y, r1a.z, r1a.w, r1b.x, r1b.y, r1b.z, r1b.w};
#pragma unroll
    for (int i = 0; i < 16; ++i) {
        float f = RELU ? fmaxf(v[i], 0.0f) : v[i];
        __bf16 hi = (__bf16)f;
        ah[i] = hi;
        al[i] = (__bf16)(f - (float)hi);
    }
}

// B-operand fragment: 32x16 bf16, lane<16 holds K 0..15, lane>=16 K 16..31 of
// column (lane&15). Source is a column-major bf16 plane -> contiguous 16 elems.
__device__ __forceinline__ v16bf load_b_frag(const __bf16* __restrict__ colptr,
                                             int kstart) {
    v8bf r0 = *(const v8bf*)(colptr + kstart);
    v8bf r1 = *(const v8bf*)(colptr + kstart + 8);
    v16bf b;
#pragma unroll
    for (int i = 0; i < 8; ++i) { b[i] = r0[i]; b[8 + i] = r1[i]; }
    return b;
}

// 3-WMMA split-precision accumulate: near-f32 product from bf16 hi/lo pairs.
__device__ __forceinline__ v8f wmma3(v8f acc, v16bf ah, v16bf al,
                                     v16bf bh, v16bf bl) {
    acc = __builtin_amdgcn_wmma_f32_16x16x32_bf16(false, ah, false, bh,
                                                  (short)0, acc, false, false);
    acc = __builtin_amdgcn_wmma_f32_16x16x32_bf16(false, ah, false, bl,
                                                  (short)0, acc, false, false);
    acc = __builtin_amdgcn_wmma_f32_16x16x32_bf16(false, al, false, bh,
                                                  (short)0, acc, false, false);
    return acc;
}

// ---------------------------------------------------------------------------
// deg / dinv:  dinv[i] = rsqrt(1 + sum_j relu(A[i,j]))   (A symmetric, diag 0)
// ---------------------------------------------------------------------------
__global__ __launch_bounds__(256) void deg_dinv_kernel(
    const float* __restrict__ adj, float* __restrict__ dinv) {
    const int row = blockIdx.x;
    const float4* a = (const float4*)(adj + (size_t)row * NN);
    float s = 0.0f;
    for (int k = threadIdx.x; k < NN / 4; k += 256) {
        float4 v = a[k];
        s += fmaxf(v.x, 0.f) + fmaxf(v.y, 0.f) + fmaxf(v.z, 0.f) + fmaxf(v.w, 0.f);
    }
    __shared__ float red[256];
    red[threadIdx.x] = s;
    __syncthreads();
    for (int off = 128; off > 0; off >>= 1) {
        if (threadIdx.x < off) red[threadIdx.x] += red[threadIdx.x + off];
        __syncthreads();
    }
    if (threadIdx.x == 0) dinv[row] = rsqrtf(red[0] + 1.0f);
}

// ---------------------------------------------------------------------------
// weight prep: W[k][n] (row-major f32) -> WT_hi/lo[n][k] (col-major bf16 planes)
// ---------------------------------------------------------------------------
__global__ __launch_bounds__(256) void wsplit_kernel(
    const float* __restrict__ W, __bf16* __restrict__ WTh,
    __bf16* __restrict__ WTl) {
    int idx = blockIdx.x * 256 + threadIdx.x;   // over 512*512
    int k = idx / HD, n = idx % HD;
    __bf16 hi, lo;
    split_bf16(W[idx], hi, lo);
    WTh[(size_t)n * HD + k] = hi;
    WTl[(size_t)n * HD + k] = lo;
}

// ---------------------------------------------------------------------------
// CONV GEMM:  Bs = dinv ⊙ (X @ W),  output split bf16 hi/lo, column-major [HD][NN]
// block: 256 thr / 8 waves, 16 rows x 512 cols (4 tiles per wave). grid: NN/16
// ---------------------------------------------------------------------------
__global__ __launch_bounds__(256) void conv_gemm_kernel(
    const float* __restrict__ X, const __bf16* __restrict__ WTh,
    const __bf16* __restrict__ WTl, const float* __restrict__ dinv,
    __bf16* __restrict__ BTh, __bf16* __restrict__ BTl) {
    const int lane = threadIdx.x & 31;
    const int wave = threadIdx.x >> 5;
    const int ln = lane & 15;
    const int half = lane >> 4;
    const int rbase = blockIdx.x * 16;
    const float* arow = X + (size_t)(rbase + ln) * HD;

    v8f acc[4] = {};
    for (int kb = 0; kb < HD; kb += 32) {
        v16bf ah, al;
        load_a_frag<false>(arow, kb + half * 16, ah, al);
        const int bk = kb + half * 16;
#pragma unroll
        for (int t = 0; t < 4; ++t) {
            const int col = t * 128 + wave * 16 + ln;
            v16bf bh = load_b_frag(WTh + (size_t)col * HD, bk);
            v16bf bl = load_b_frag(WTl + (size_t)col * HD, bk);
            acc[t] = wmma3(acc[t], ah, al, bh, bl);
        }
    }
    // epilogue: scale rows by dinv, split, transposed b128 stores
    const float* dv = dinv + rbase + half * 8;
    float d[8];
#pragma unroll
    for (int j = 0; j < 8; ++j) d[j] = dv[j];
#pragma unroll
    for (int t = 0; t < 4; ++t) {
        const int col = t * 128 + wave * 16 + ln;
        v8bf oh, ol;
#pragma unroll
        for (int j = 0; j < 8; ++j) {
            __bf16 hi, lo;
            split_bf16(acc[t][j] * d[j], hi, lo);
            oh[j] = hi;
            ol[j] = lo;
        }
        const size_t off = (size_t)col * NN + rbase + half * 8;
        *(v8bf*)(BTh + off) = oh;
        *(v8bf*)(BTl + off) = ol;
    }
}

// ---------------------------------------------------------------------------
// PROP:  Z = dinv ⊙ ((A+I) @ Bs) + conv_b,  A streamed f32, relu+split on the fly
// block: 256 thr / 8 waves, 16 rows x 256 cols (2 tiles/wave). grid: (2, NN/16)
// ---------------------------------------------------------------------------
__global__ __launch_bounds__(256) void prop_kernel(
    const float* __restrict__ adj, const __bf16* __restrict__ BTh,
    const __bf16* __restrict__ BTl, const float* __restrict__ dinv,
    const float* __restrict__ bias, float* __restrict__ out) {
    const int lane = threadIdx.x & 31;
    const int wave = threadIdx.x >> 5;
    const int ln = lane & 15;
    const int half = lane >> 4;
    const int rbase = blockIdx.y * 16;
    const int cbase = blockIdx.x * 256;
    const float* arow = adj + (size_t)(rbase + ln) * NN;

    v8f acc[2] = {};
    for (int kb = 0; kb < NN; kb += 32) {
        __builtin_prefetch(arow + kb + 1024, 0, 0);   // global_prefetch_b8
        v16bf ah, al;
        load_a_frag<true>(arow, kb + half * 16, ah, al);
        const int bk = kb + half * 16;
#pragma unroll
        for (int t = 0; t < 2; ++t) {
            const int col = cbase + t * 128 + wave * 16 + ln;
            v16bf bh = load_b_frag(BTh + (size_t)col * NN, bk);
            v16bf bl = load_b_frag(BTl + (size_t)col * NN, bk);
            acc[t] = wmma3(acc[t], ah, al, bh, bl);
        }
    }
    // epilogue: self-loop term + row scale + bias, row-major f32 store
    const float* dv = dinv + rbase + half * 8;
    float d[8];
#pragma unroll
    for (int j = 0; j < 8; ++j) d[j] = dv[j];
#pragma unroll
    for (int t = 0; t < 2; ++t) {
        const int col = cbase + t * 128 + wave * 16 + ln;
        const size_t soff = (size_t)col * NN + rbase + half * 8;
        v8bf sh = *(const v8bf*)(BTh + soff);
        v8bf sl = *(const v8bf*)(BTl + soff);
        const float b = bias[col];
#pragma unroll
        for (int j = 0; j < 8; ++j) {
            const float self = (float)sh[j] + (float)sl[j];
            const int row = rbase + j + 8 * half;
            out[(size_t)row * HD + col] = d[j] * (acc[t][j] + self) + b;
        }
    }
}

// ---------------------------------------------------------------------------
// MLP / FINAL GEMM:  Y = X @ W + b; optionally LayerNorm(g,be)+ReLU fused via LDS
// block: 256 thr / 8 waves, 16 rows x 512 cols. grid: NN/16
// ---------------------------------------------------------------------------
__global__ __launch_bounds__(256) void mlp_gemm_kernel(
    const float* __restrict__ X, const __bf16* __restrict__ WTh,
    const __bf16* __restrict__ WTl, const float* __restrict__ bias,
    const float* __restrict__ g, const float* __restrict__ be,
    float* __restrict__ out, int do_ln) {
    const int lane = threadIdx.x & 31;
    const int wave = threadIdx.x >> 5;
    const int ln = lane & 15;
    const int half = lane >> 4;
    const int rbase = blockIdx.x * 16;
    const float* arow = X + (size_t)(rbase + ln) * HD;

    v8f acc[4] = {};
    for (int kb = 0; kb < HD; kb += 32) {
        v16bf ah, al;
        load_a_frag<false>(arow, kb + half * 16, ah, al);
        const int bk = kb + half * 16;
#pragma unroll
        for (int t = 0; t < 4; ++t) {
            const int col = t * 128 + wave * 16 + ln;
            v16bf bh = load_b_frag(WTh + (size_t)col * HD, bk);
            v16bf bl = load_b_frag(WTl + (size_t)col * HD, bk);
            acc[t] = wmma3(acc[t], ah, al, bh, bl);
        }
    }

    if (!do_ln) {  // final linear: bias add, row-major store
#pragma unroll
        for (int t = 0; t < 4; ++t) {
            const int col = t * 128 + wave * 16 + ln;
            const float b = bias[col];
#pragma unroll
            for (int j = 0; j < 8; ++j) {
                const int row = rbase + j + 8 * half;
                out[(size_t)row * HD + col] = acc[t][j] + b;
            }
        }
        return;
    }

    // fused LayerNorm + ReLU (LDS stride 516 f32 -> conflict-free tile writes)
    __shared__ float lnbuf[16 * 516];
    __shared__ float psum[256], psq[256];
    __shared__ float mu_s[16], rs_s[16];
#pragma unroll
    for (int t = 0; t < 4; ++t) {
        const int col = t * 128 + wave * 16 + ln;
        const float b = bias[col];
#pragma unroll
        for (int j = 0; j < 8; ++j)
            lnbuf[(j + 8 * half) * 516 + col] = acc[t][j] + b;
    }
    __syncthreads();
    const int r = threadIdx.x >> 4;
    const int seg = threadIdx.x & 15;
    float s = 0.f, sq = 0.f;
#pragma unroll
    for (int i = 0; i < 32; ++i) {
        const float v = lnbuf[r * 516 + seg * 32 + i];
        s += v;
        sq += v * v;
    }
    psum[r * 16 + seg] = s;
    psq[r * 16 + seg] = sq;
    __syncthreads();
    if (threadIdx.x < 16) {
        float S = 0.f, Q = 0.f;
#pragma unroll
        for (int i = 0; i < 16; ++i) {
            S += psum[threadIdx.x * 16 + i];
            Q += psq[threadIdx.x * 16 + i];
        }
        const float mu = S * (1.0f / HD);
        const float var = Q * (1.0f / HD) - mu * mu;
        mu_s[threadIdx.x] = mu;
        rs_s[threadIdx.x] = rsqrtf(var + LN_EPS);
    }
    __syncthreads();
    const float mu = mu_s[r], rs = rs_s[r];
    float* orow = out + (size_t)(rbase + r) * HD;
#pragma unroll
    for (int i = 0; i < 32; ++i) {
        const int col = seg * 32 + i;
        const float v = lnbuf[r * 516 + col];
        orow[col] = fmaxf((v - mu) * rs * g[col] + be[col], 0.0f);
    }
}

// ---------------------------------------------------------------------------
// launch
// ---------------------------------------------------------------------------
extern "C" void kernel_launch(void* const* d_in, const int* in_sizes, int n_in,
                              void* d_out, int out_size, void* d_ws,
                              size_t ws_size, hipStream_t stream) {
    (void)in_sizes; (void)n_in; (void)out_size; (void)ws_size;
    const float* node_feat = (const float*)d_in[0];
    const float* adj       = (const float*)d_in[1];
    const float* conv_w    = (const float*)d_in[2];
    const float* conv_b    = (const float*)d_in[3];
    const float* mlp_w     = (const float*)d_in[4];
    const float* mlp_b     = (const float*)d_in[5];
    const float* ln_g      = (const float*)d_in[6];
    const float* ln_b      = (const float*)d_in[7];
    const float* lin_w     = (const float*)d_in[8];
    const float* lin_b     = (const float*)d_in[9];

    char* p = (char*)d_ws;
    auto carve = [&](size_t bytes) {
        void* r = (void*)p;
        p += (bytes + 255) & ~(size_t)255;
        return r;
    };
    float*  dinv = (float*)carve(NN * sizeof(float));
    __bf16* BTh  = (__bf16*)carve((size_t)HD * NN * 2);
    __bf16* BTl  = (__bf16*)carve((size_t)HD * NN * 2);
    __bf16* WTh[7];
    __bf16* WTl[7];
    for (int i = 0; i < 7; ++i) {
        WTh[i] = (__bf16*)carve((size_t)HD * HD * 2);
        WTl[i] = (__bf16*)carve((size_t)HD * HD * 2);
    }
    float* xA = (float*)carve((size_t)NN * HD * sizeof(float));
    float* xB = (float*)carve((size_t)NN * HD * sizeof(float));

    const size_t HH = (size_t)HD * HD;
    const float* Wsrc[7] = {conv_w, conv_w + HH, conv_w + 2 * HH,
                            mlp_w,  mlp_w + HH,  mlp_w + 2 * HH, lin_w};

    deg_dinv_kernel<<<NN, 256, 0, stream>>>(adj, dinv);
    for (int i = 0; i < 7; ++i)
        wsplit_kernel<<<(HD * HD) / 256, 256, 0, stream>>>(Wsrc[i], WTh[i], WTl[i]);

    const float* x = node_feat;
    for (int l = 0; l < NLAY; ++l) {
        conv_gemm_kernel<<<NN / 16, 256, 0, stream>>>(x, WTh[l], WTl[l], dinv,
                                                      BTh, BTl);
        prop_kernel<<<dim3(2, NN / 16), 256, 0, stream>>>(
            adj, BTh, BTl, dinv, conv_b + l * HD, xA);
        mlp_gemm_kernel<<<NN / 16, 256, 0, stream>>>(
            xA, WTh[3 + l], WTl[3 + l], mlp_b + l * HD, ln_g + l * HD,
            ln_b + l * HD, xB, 1);
        x = xB;
    }
    // final linear: reuse MLP kernel with LN disabled
    mlp_gemm_kernel<<<NN / 16, 256, 0, stream>>>(x, WTh[6], WTl[6], lin_b,
                                                 ln_g, ln_b, (float*)d_out, 0);
}
